// GCL_29858612642363
// MI455X (gfx1250) — matrix operand order repaired and checked by reading, not compiled, not run on hardware
//
#include <hip/hip_runtime.h>

#define NN  50000
#define EE  800000
#define NF  128
#define HID 256

typedef __bf16 v4bf  __attribute__((ext_vector_type(4)));
typedef __bf16 v8bf  __attribute__((ext_vector_type(8)));
typedef __bf16 v16bf __attribute__((ext_vector_type(16)));
typedef float  v8f   __attribute__((ext_vector_type(8)));

// int4 vector + address-space-qualified pointer types for the async builtin.
typedef int i32x4 __attribute__((ext_vector_type(4)));
typedef __attribute__((address_space(1))) i32x4* gas1_i4p;   // global (AS1)
typedef __attribute__((address_space(3))) i32x4* las3_i4p;   // LDS (AS3)

__device__ __forceinline__ v16bf bf_combine(v8bf lo, v8bf hi) {
    return __builtin_shufflevector(lo, hi, 0,1,2,3,4,5,6,7,8,9,10,11,12,13,14,15);
}

__device__ __forceinline__ v8f wmma_bf16(v16bf a, v16bf b, v8f c) {
    // 8-arg form: (neg_a, A, neg_b, B, c_mod, C, reuse_a, reuse_b)
    return __builtin_amdgcn_wmma_f32_16x16x32_bf16(false, a, false, b, (short)0, c, false, false);
}

// Async global->LDS 16B copy (per lane), tracked by ASYNCcnt.
// ldsOff is the 32-bit LDS byte address (low 32 bits of the generic pointer).
__device__ __forceinline__ void async_ldg_b128(const void* g, unsigned ldsOff) {
#if __has_builtin(__builtin_amdgcn_global_load_async_to_lds_b128)
    __builtin_amdgcn_global_load_async_to_lds_b128(
        (gas1_i4p)(uintptr_t)g,
        (las3_i4p)(uintptr_t)ldsOff,
        0, 0);
#else
    asm volatile("global_load_async_to_lds_b128 %0, %1, off"
                 :: "v"(ldsOff), "v"((unsigned long long)(uintptr_t)g)
                 : "memory");
#endif
}

__device__ __forceinline__ void wait_asynccnt0() {
#if __has_builtin(__builtin_amdgcn_s_wait_asynccnt)
    __builtin_amdgcn_s_wait_asynccnt(0);
#else
    asm volatile("s_wait_asynccnt 0x0" ::: "memory");
#endif
}

// ---------------------------------------------------------------------------
// Pack a row-major fp32 weight [K x N] into bf16 WMMA B-fragments.
// B 32x16 bf16 layout (ISA 7.12.2): lanes 0-15 hold column n = nt*16+lane with
// K = kt*32 + {0..7,16..23}; lanes 16-31 hold K = kt*32 + {8..15,24..31}.
// Fragment = 512 bf16 (1 KB), lane-major -> one 32B load per lane.
// ---------------------------------------------------------------------------
__global__ void pack_w_kernel(const float* __restrict__ W, __bf16* __restrict__ dst,
                              int K, int N) {
    int tid = blockIdx.x * 256 + threadIdx.x;
    if (tid >= K * N) return;
    int frag   = tid >> 9;
    int within = tid & 511;
    int lane   = within >> 4;
    int j      = within & 15;
    int NT = N >> 4;
    int kt = frag / NT;
    int nt = frag - kt * NT;
    int kl = (j & 7) + ((lane >> 4) << 3) + ((j >> 3) << 4);
    int k  = kt * 32 + kl;
    int n  = nt * 16 + (lane & 15);
    dst[tid] = (__bf16)W[(long)k * N + n];
}

// A-fragment load from row-major bf16 LDS tile.
// A 16x32 layout: lanes 0-15 row m hold K {0..7,16..23}; lanes 16-31 hold
// K {8..15,24..31} -> two contiguous 16B LDS loads.
__device__ __forceinline__ v16bf load_a(const __bf16* sA, int rowStride,
                                        int m, int kt, int hi8) {
    const __bf16* p = sA + m * rowStride + kt * 32 + hi8;
    return bf_combine(*(const v8bf*)p, *(const v8bf*)(p + 16));
}

// ---------------------------------------------------------------------------
// Edge kernel: 256 edges / block, 8 waves; wave w owns M-tiles {2w, 2w+1}
// (rows w*32..w*32+31) so every B fragment feeds two WMMAs.
// W2 fragments are async-staged into LDS during gather+GEMM1.
// ---------------------------------------------------------------------------
__global__ __launch_bounds__(256)
void egnn_edge_kernel(const float* __restrict__ h,
                      const int*   __restrict__ eidx,
                      const __bf16* __restrict__ w1p,
                      const float* __restrict__ b1,
                      const float* __restrict__ g1,
                      const float* __restrict__ be1,
                      const __bf16* __restrict__ w2p,
                      const float* __restrict__ b2,
                      float* __restrict__ agg) {
    extern __shared__ char smem[];
    __bf16* sA    = (__bf16*)smem;                    // [256][256] bf16, 128 KB
    __bf16* sW2   = (__bf16*)(smem + 131072);         // packed W2, 128 KB
    int*    rowid = (int*)(smem + 262144);            // [256]

    const int  tid   = threadIdx.x;
    const int  lane  = tid & 31;
    const int  w     = tid >> 5;
    const long eBase = (long)blockIdx.x * 256;

    // Kick off async staging of W2 fragments into LDS (overlaps gather+GEMM1).
    {
        const char* gW2 = (const char*)w2p;
        unsigned ldsBase = (unsigned)(uintptr_t)sW2;
#pragma unroll
        for (int i = 0; i < 32; ++i) {
            int off = ((w * 32 + i) * 32 + lane) * 16;   // 256 x 512B chunks
            async_ldg_b128(gW2 + off, ldsBase + off);
        }
    }

    rowid[tid] = eidx[eBase + tid];

    // Gather e_in = [h[row] | h[col]] -> bf16 LDS tile (h is L2-resident).
    {
        int r0 = eidx[eBase + tid];
        int c0 = eidx[(long)EE + eBase + tid];
        const float4* s0 = (const float4*)(h + (long)r0 * NF);
        const float4* s1 = (const float4*)(h + (long)c0 * NF);
        __bf16* dst = sA + tid * 256;
#pragma unroll
        for (int i = 0; i < 32; ++i) {
            float4 a = s0[i], b = s1[i];
            v4bf pa = {(__bf16)a.x, (__bf16)a.y, (__bf16)a.z, (__bf16)a.w};
            v4bf pb = {(__bf16)b.x, (__bf16)b.y, (__bf16)b.z, (__bf16)b.w};
            *(v4bf*)(dst + i * 4)       = pa;
            *(v4bf*)(dst + 128 + i * 4) = pb;
        }
    }
    __syncthreads();

    const int ml  = lane & 15;
    const int hi8 = (lane >> 4) << 3;
    const int m0a = w * 32 + ml;            // first M-tile row
    const int m0b = w * 32 + 16 + ml;       // second M-tile row

    // GEMM1: [256x256] @ W1 ; B fragment reused across both M-tiles.
    v8f acc[2][16] = {};
    for (int kt = 0; kt < 8; ++kt) {
        v16bf a0 = load_a(sA, 256, m0a, kt, hi8);
        v16bf a1 = load_a(sA, 256, m0b, kt, hi8);
        const __bf16* pb = w1p + (long)(kt * 16) * 512 + lane * 16;
#pragma unroll
        for (int nt = 0; nt < 16; ++nt) {
            v16bf b = *(const v16bf*)(pb + nt * 512);
            acc[0][nt] = wmma_bf16(a0, b, acc[0][nt]);
            acc[1][nt] = wmma_bf16(a1, b, acc[1][nt]);
        }
    }

    // bias + LayerNorm (half-wave shfl matches C layout) + SiLU -> back to LDS
    float b1n[16], g1n[16], e1n[16];
#pragma unroll
    for (int nt = 0; nt < 16; ++nt) {
        int n = nt * 16 + ml;
        b1n[nt] = b1[n]; g1n[nt] = g1[n]; e1n[nt] = be1[n];
    }
#pragma unroll
    for (int t = 0; t < 2; ++t) {
#pragma unroll
        for (int r = 0; r < 8; ++r) {
            float s = 0.f, q = 0.f;
#pragma unroll
            for (int nt = 0; nt < 16; ++nt) {
                float x = acc[t][nt][r] + b1n[nt];
                s += x; q += x * x;
            }
#pragma unroll
            for (int off = 1; off < 16; off <<= 1) {
                s += __shfl_xor(s, off, 16);
                q += __shfl_xor(q, off, 16);
            }
            float mu   = s * (1.f / HID);
            float rstd = rsqrtf(q * (1.f / HID) - mu * mu + 1e-5f);
            int m = w * 32 + t * 16 + r + hi8;   // C layout: upper lanes row r+8
#pragma unroll
            for (int nt = 0; nt < 16; ++nt) {
                float x = acc[t][nt][r] + b1n[nt];
                float y = (x - mu) * rstd * g1n[nt] + e1n[nt];
                y = y / (1.f + __expf(-y));      // SiLU
                sA[m * 256 + nt * 16 + ml] = (__bf16)y;  // rows are wave-private
            }
        }
    }

    // W2 fragments must be resident in LDS for GEMM2.
    wait_asynccnt0();
    __syncthreads();

    // GEMM2: silu(LN(x)) @ W2, B fragments from LDS.
    v8f acc2[2][16] = {};
    for (int kt = 0; kt < 8; ++kt) {
        v16bf a0 = load_a(sA, 256, m0a, kt, hi8);
        v16bf a1 = load_a(sA, 256, m0b, kt, hi8);
        const __bf16* pb = sW2 + (long)(kt * 16) * 512 + lane * 16;
#pragma unroll
        for (int nt = 0; nt < 16; ++nt) {
            v16bf b = *(const v16bf*)(pb + nt * 512);
            acc2[0][nt] = wmma_bf16(a0, b, acc2[0][nt]);
            acc2[1][nt] = wmma_bf16(a1, b, acc2[1][nt]);
        }
    }

    // +b2, scatter-add into agg[row[e]]
#pragma unroll
    for (int nt = 0; nt < 16; ++nt) {
        int n = nt * 16 + ml;
        float bb = b2[n];
#pragma unroll
        for (int t = 0; t < 2; ++t) {
#pragma unroll
            for (int r = 0; r < 8; ++r) {
                int m = w * 32 + t * 16 + r + hi8;
                int node = rowid[m];
                atomicAdd(agg + (long)node * HID + n, acc2[t][nt][r] + bb);
            }
        }
    }
}

// ---------------------------------------------------------------------------
// Node kernel: 128 nodes / block; K=384 concat GEMM, LN, SiLU, 256->128 GEMM,
// residual. Tail rows clamped on gather, guarded on store (WMMA keeps full EXEC).
// ---------------------------------------------------------------------------
__global__ __launch_bounds__(256)
void egnn_node_kernel(const float* __restrict__ h,
                      const float* __restrict__ agg,
                      const __bf16* __restrict__ w3p,
                      const float* __restrict__ b3,
                      const float* __restrict__ g2,
                      const float* __restrict__ be2,
                      const __bf16* __restrict__ w4p,
                      const float* __restrict__ b4,
                      float* __restrict__ out) {
    extern __shared__ char smem[];
    __bf16* sA = (__bf16*)smem;                      // [128][384]
    __bf16* sX = (__bf16*)(smem + 128 * 384 * 2);    // [128][256]

    const int tid   = threadIdx.x;
    const int lane  = tid & 31;
    const int w     = tid >> 5;
    const int nBase = blockIdx.x * 128;

    for (int idx = tid; idx < 128 * 96; idx += 256) {
        int nl = idx / 96;
        int c  = idx - nl * 96;
        int node = nBase + nl; if (node >= NN) node = NN - 1;
        float4 v = (c < 32) ? ((const float4*)(h   + (long)node * NF ))[c]
                            : ((const float4*)(agg + (long)node * HID))[c - 32];
        v4bf p = {(__bf16)v.x, (__bf16)v.y, (__bf16)v.z, (__bf16)v.w};
        *(v4bf*)(sA + nl * 384 + c * 4) = p;
    }
    __syncthreads();

    const int ml  = lane & 15;
    const int hi8 = (lane >> 4) << 3;
    const int m0  = w * 16 + ml;

    // GEMM3: [128x384] @ W3
    v8f acc[16] = {};
    for (int kt = 0; kt < 12; ++kt) {
        v16bf a = load_a(sA, 384, m0, kt, hi8);
        const __bf16* pb = w3p + (long)(kt * 16) * 512 + lane * 16;
#pragma unroll
        for (int nt = 0; nt < 16; ++nt) {
            v16bf b = *(const v16bf*)(pb + nt * 512);
            acc[nt] = wmma_bf16(a, b, acc[nt]);
        }
    }

    float bn[16], gn[16], en[16];
#pragma unroll
    for (int nt = 0; nt < 16; ++nt) {
        int n = nt * 16 + ml;
        bn[nt] = b3[n]; gn[nt] = g2[n]; en[nt] = be2[n];
    }
#pragma unroll
    for (int r = 0; r < 8; ++r) {
        float s = 0.f, q = 0.f;
#pragma unroll
        for (int nt = 0; nt < 16; ++nt) {
            float x = acc[nt][r] + bn[nt];
            s += x; q += x * x;
        }
#pragma unroll
        for (int off = 1; off < 16; off <<= 1) {
            s += __shfl_xor(s, off, 16);
            q += __shfl_xor(q, off, 16);
        }
        float mu   = s * (1.f / HID);
        float rstd = rsqrtf(q * (1.f / HID) - mu * mu + 1e-5f);
        int m = w * 16 + r + hi8;
#pragma unroll
        for (int nt = 0; nt < 16; ++nt) {
            float x = acc[nt][r] + bn[nt];
            float y = (x - mu) * rstd * gn[nt] + en[nt];
            y = y / (1.f + __expf(-y));
            sX[m * 256 + nt * 16 + ml] = (__bf16)y;
        }
    }

    // GEMM4: [128x256] @ W4 (Nout = 128 -> 8 N-tiles)
    v8f acc2[8] = {};
    for (int kt = 0; kt < 8; ++kt) {
        v16bf a = load_a(sX, 256, m0, kt, hi8);
        const __bf16* pb = w4p + (long)(kt * 8) * 512 + lane * 16;
#pragma unroll
        for (int nt = 0; nt < 8; ++nt) {
            v16bf b = *(const v16bf*)(pb + nt * 512);
            acc2[nt] = wmma_bf16(a, b, acc2[nt]);
        }
    }

#pragma unroll
    for (int nt = 0; nt < 8; ++nt) {
        int n = nt * 16 + ml;
        float bb = b4[n];
#pragma unroll
        for (int r = 0; r < 8; ++r) {
            int m = w * 16 + r + hi8;
            int node = nBase + m;
            if (node < NN)
                out[(long)node * NF + n] = h[(long)node * NF + n] + acc2[nt][r] + bb;
        }
    }
}

// ---------------------------------------------------------------------------
extern "C" void kernel_launch(void* const* d_in, const int* in_sizes, int n_in,
                              void* d_out, int out_size, void* d_ws, size_t ws_size,
                              hipStream_t stream) {
    (void)in_sizes; (void)n_in; (void)out_size; (void)ws_size;
    const float* h    = (const float*)d_in[0];
    const int*   eidx = (const int*)  d_in[1];
    const float* W1   = (const float*)d_in[2];
    const float* b1   = (const float*)d_in[3];
    const float* g1   = (const float*)d_in[4];
    const float* be1  = (const float*)d_in[5];
    const float* W2   = (const float*)d_in[6];
    const float* b2   = (const float*)d_in[7];
    const float* W3   = (const float*)d_in[8];
    const float* b3   = (const float*)d_in[9];
    const float* g2   = (const float*)d_in[10];
    const float* be2  = (const float*)d_in[11];
    const float* W4   = (const float*)d_in[12];
    const float* b4   = (const float*)d_in[13];
    float* out = (float*)d_out;

    char* ws = (char*)d_ws;
    float*  agg = (float*)ws;                       // 50000*256*4 = 51,200,000 B
    size_t  off = (size_t)NN * HID * 4;
    __bf16* w1p = (__bf16*)(ws + off); off += 256 * 256 * 2;
    __bf16* w2p = (__bf16*)(ws + off); off += 256 * 256 * 2;
    __bf16* w3p = (__bf16*)(ws + off); off += 384 * 256 * 2;
    __bf16* w4p = (__bf16*)(ws + off);

    (void)hipMemsetAsync(agg, 0, (size_t)NN * HID * 4, stream);

    pack_w_kernel<<<256, 256, 0, stream>>>(W1, w1p, 256, 256);
    pack_w_kernel<<<256, 256, 0, stream>>>(W2, w2p, 256, 256);
    pack_w_kernel<<<384, 256, 0, stream>>>(W3, w3p, 384, 256);
    pack_w_kernel<<<128, 256, 0, stream>>>(W4, w4p, 256, 128);

    const size_t edgeLds = 256 * 256 * 2 + 256 * 256 * 2 + 256 * 4;  // 263,168 B
    egnn_edge_kernel<<<EE / 256, 256, edgeLds, stream>>>(
        h, eidx, w1p, b1, g1, be1, w2p, b2, agg);

    const size_t nodeLds = 128 * 384 * 2 + 128 * 256 * 2;            // 163,840 B
    egnn_node_kernel<<<(NN + 127) / 128, 256, nodeLds, stream>>>(
        h, agg, w3p, b3, g2, be2, w4p, b4, out);
}
